// CausalTensionGraphLayer_85177791414734
// MI455X (gfx1250) — matrix operand order repaired and checked by reading, not compiled, not run on hardware
//
#include <hip/hip_runtime.h>
#include <hip/hip_bf16.h>
#include <math.h>

// ---------------------------------------------------------------------------
// CausalTensionGraphLayer:
//   GEMM1: C1[8192,2048] = x @ [w1_top | w1_bot | wv_w]   (bf16 WMMA, LDS async)
//   Pass2: tau gates + msg accumulation
//   GEMM3: y = [x | msg] @ merge_w + merge_b              (bf16 WMMA, LDS async)
//   LN   : in-place row LayerNorm on d_out
// GEMM operands pre-packed into WMMA fragment-native layouts; k-step tiles are
// double-buffered in LDS via GLOBAL_LOAD_ASYNC_TO_LDS_B128 (ASYNCcnt).
// ---------------------------------------------------------------------------

typedef __attribute__((ext_vector_type(16))) __bf16 v16bf;
typedef __attribute__((ext_vector_type(8)))  float  v8f;
typedef __attribute__((ext_vector_type(4)))  int    i32x4;

#define Bn   2
#define Tn   4096
#define Dn   1024
#define Hn   512
#define BT   8192
#define N1   2048
#define KT1  32
#define NT1  128
#define KT3  64
#define NT3  64

#if defined(__has_builtin)
#if __has_builtin(__builtin_amdgcn_global_load_async_to_lds_b128) && \
    __has_builtin(__builtin_amdgcn_s_wait_asynccnt)
#define USE_ASYNC_LDS 1
#endif
#endif

// Copy 16 bytes global -> LDS (async when available).
__device__ __forceinline__ void cp16(__bf16* dst, const __bf16* src) {
#ifdef USE_ASYNC_LDS
  __builtin_amdgcn_global_load_async_to_lds_b128(
      (__attribute__((address_space(1))) i32x4*)src,
      (__attribute__((address_space(3))) i32x4*)dst, 0, 0);
#else
  *(uint4*)dst = *(const uint4*)src;
#endif
}

__device__ __forceinline__ void async_wait4() {
#ifdef USE_ASYNC_LDS
  __builtin_amdgcn_s_wait_asynccnt(4);   // allow next-stage (4 ops/wave) in flight
#endif
}

// A-matrix (16x32 bf16) per-lane fragment layout (ISA 7.12.2).
__device__ __forceinline__ void frag_a(int lane, int j, int& m, int& k) {
  m = lane & 15;
  int kb = (lane >> 4) * 8;
  int p = j >> 1, s = j & 1;
  k = (p < 4) ? (kb + 2 * p + s) : (16 + kb + 2 * (p - 4) + s);
}

// B-matrix (32x16 bf16) per-lane fragment layout.
__device__ __forceinline__ void frag_b(int lane, int j, int& n, int& k) {
  n = lane & 15;
  k = ((lane >> 4) << 3) + ((j < 8) ? j : (j + 8));
}

// Stage one k-step tile into LDS: A = 8 strips x 512 elems (stride astride),
// B = 8 contiguous frags (4096 elems). 4 x 16B ops per thread.
__device__ __forceinline__ void stage_tile(__bf16* dst, const __bf16* asrc,
                                           long astride, const __bf16* bsrc,
                                           int tid) {
#pragma unroll
  for (int i = 0; i < 2; ++i) {
    int u = tid + i * 256;                 // 512 A units
    int strip = u >> 6, off = (u & 63) * 8;
    cp16(dst + strip * 512 + off, asrc + (long)strip * astride + off);
  }
#pragma unroll
  for (int i = 0; i < 2; ++i) {
    int u = tid + i * 256;                 // 512 B units, contiguous
    cp16(dst + 4096 + u * 8, bsrc + u * 8);
  }
}

// --------------------------- packing kernels -------------------------------

__global__ __launch_bounds__(256) void k_pack_x(const float* __restrict__ x,
                                                __bf16* __restrict__ xp) {
  long e = (long)blockIdx.x * 256 + threadIdx.x;   // < 512*32*512
  int j = e & 15, lane = (e >> 4) & 31, kt = (e >> 9) & 31, mt = (int)(e >> 14);
  int m, k; frag_a(lane, j, m, k);
  xp[e] = (__bf16)x[(long)(mt * 16 + m) * Dn + kt * 32 + k];
}

__global__ __launch_bounds__(256) void k_pack_w(const float* __restrict__ w1,
                                                const float* __restrict__ wv,
                                                __bf16* __restrict__ wp) {
  long e = (long)blockIdx.x * 256 + threadIdx.x;   // < 32*128*512
  int j = e & 15, lane = (e >> 4) & 31, nt = (e >> 9) & 127, kt = (int)(e >> 16);
  int n, k; frag_b(lane, j, n, k);
  int ng = nt * 16 + n, kg = kt * 32 + k;
  float v;
  if (ng < 512)        v = w1[(long)kg * Hn + ng];
  else if (ng < 1024)  v = w1[(long)(Dn + kg) * Hn + (ng - 512)];
  else                 v = wv[(long)kg * Dn + (ng - 1024)];
  wp[e] = (__bf16)v;
}

__global__ __launch_bounds__(256) void k_pack_mw(const float* __restrict__ mw,
                                                 __bf16* __restrict__ wp) {
  long e = (long)blockIdx.x * 256 + threadIdx.x;   // < 64*64*512
  int j = e & 15, lane = (e >> 4) & 31, nt = (e >> 9) & 63, kt = (int)(e >> 15);
  int n, k; frag_b(lane, j, n, k);
  wp[e] = (__bf16)mw[(long)(kt * 32 + k) * Dn + nt * 16 + n];
}

// ------------------------------- GEMM1 -------------------------------------
// grid (64,16), 256 thr (8 waves). Block tile 128x128; double-buffered LDS.
__global__ __launch_bounds__(256) void k_gemm1(const __bf16* __restrict__ xp,
                                               const __bf16* __restrict__ wp,
                                               __bf16* __restrict__ c1) {
  __shared__ __bf16 sm[2][8192];           // [buf][A 4096 | B 4096] = 32 KB
  int tid = threadIdx.x, wave = tid >> 5, lane = tid & 31;
  int mtile0 = blockIdx.x * 8;
  int nt0    = blockIdx.y * 8;
  const __bf16* aR = xp + (long)mtile0 * KT1 * 512;
  const __bf16* bR = wp + (long)nt0 * 512;

  v8f acc[8];
  v8f zero = {0.f, 0.f, 0.f, 0.f, 0.f, 0.f, 0.f, 0.f};
#pragma unroll
  for (int t = 0; t < 8; ++t) acc[t] = zero;

  stage_tile(sm[0], aR, (long)KT1 * 512, bR, tid);

  for (int kt = 0; kt < KT1; ++kt) {
    int cur = kt & 1;
    if (kt + 1 < KT1)
      stage_tile(sm[cur ^ 1], aR + (long)(kt + 1) * 512, (long)KT1 * 512,
                 bR + (long)(kt + 1) * NT1 * 512, tid);
    async_wait4();                         // current buffer complete (in-order)
    __syncthreads();

    v16bf a = *(const v16bf*)(&sm[cur][wave * 512 + lane * 16]);
    v16bf bf[8];
#pragma unroll
    for (int t = 0; t < 8; ++t)
      bf[t] = *(const v16bf*)(&sm[cur][4096 + t * 512 + lane * 16]);
#pragma unroll
    for (int t = 0; t < 8; ++t)
      acc[t] = __builtin_amdgcn_wmma_f32_16x16x32_bf16(
          false, a, false, bf[t], (short)0, acc[t], false, false);

    __syncthreads();                       // WAR: readers done before rewrite
  }

  int mb = (mtile0 + wave) * 16 + ((lane >> 4) << 3);
  int n  = lane & 15;
#pragma unroll
  for (int t = 0; t < 8; ++t) {
    int col = (nt0 + t) * 16 + n;
#pragma unroll
    for (int r = 0; r < 8; ++r)
      c1[(long)(mb + r) * N1 + col] = (__bf16)acc[t][r];
  }
}

// ------------------------------- Pass 2 ------------------------------------
__global__ __launch_bounds__(256) void k_pass2(const __bf16* __restrict__ c1,
                                               const float* __restrict__ b1,
                                               const float* __restrict__ w2,
                                               const float* __restrict__ b2p,
                                               const float* __restrict__ wvb,
                                               __bf16* __restrict__ msgp) {
  __shared__ float red[32];
  __shared__ float taus[4];
  int bt = blockIdx.x;
  int b  = bt / Tn, t = bt - b * Tn;
  int tid = threadIdx.x, lane = tid & 31, wave = tid >> 5;
  long rowA = (long)bt * N1;

  float a0  = (float)c1[rowA + tid];
  float a1  = (float)c1[rowA + 256 + tid];
  float bb0 = b1[tid], bb1 = b1[256 + tid];
  float w20 = w2[tid], w21 = w2[256 + tid];

  float part[4];
#pragma unroll
  for (int w = 0; w < 4; ++w) {
    int tp = t - 1 - w;
    float bn0 = 0.f, bn1 = 0.f;
    if (tp >= 0) {
      long rowB = (long)(b * Tn + tp) * N1 + Hn;
      bn0 = (float)c1[rowB + tid];
      bn1 = (float)c1[rowB + 256 + tid];
    }
    float h0 = a0 + bn0 + bb0; h0 = h0 / (1.f + __expf(-h0));
    float h1 = a1 + bn1 + bb1; h1 = h1 / (1.f + __expf(-h1));
    part[w] = h0 * w20 + h1 * w21;
  }
#pragma unroll
  for (int w = 0; w < 4; ++w) {
    float s = part[w];
#pragma unroll
    for (int off = 16; off > 0; off >>= 1) s += __shfl_down(s, off, 32);
    if (lane == 0) red[wave * 4 + w] = s;
  }
  __syncthreads();
  if (tid < 4) {
    float s = 0.f;
    for (int wv = 0; wv < 8; ++wv) s += red[wv * 4 + tid];
    taus[tid] = 1.f / (1.f + __expf(-(s + b2p[0])));
  }
  __syncthreads();
  float tw[4] = {taus[0], taus[1], taus[2], taus[3]};
  float ts = tw[0] + tw[1] + tw[2] + tw[3];

  int mt = bt >> 4, mm = bt & 15;
#pragma unroll
  for (int i = 0; i < 4; ++i) {
    int d = tid + i * 256;
    float m = ts * wvb[d];
#pragma unroll
    for (int w = 0; w < 4; ++w) {
      int tp = t - 1 - w;
      if (tp >= 0)
        m += tw[w] * (float)c1[(long)(b * Tn + tp) * N1 + 2 * Hn + d];
    }
    int kt = d >> 5, kin = d & 31;
    int lanehi = (kin >> 3) & 1;
    int off = kin - ((kin >= 16) ? 16 : 0) - lanehi * 8;
    int p = ((kin >= 16) ? 4 : 0) + (off >> 1);
    int j = 2 * p + (off & 1);
    int fl = lanehi * 16 + mm;
    msgp[(((long)(mt * 32 + kt)) << 9) + fl * 16 + j] = (__bf16)m;
  }
}

// ------------------------------- GEMM3 -------------------------------------
__global__ __launch_bounds__(256) void k_gemm3(const __bf16* __restrict__ xp,
                                               const __bf16* __restrict__ mp,
                                               const __bf16* __restrict__ wp,
                                               const float* __restrict__ mbias,
                                               float* __restrict__ out) {
  __shared__ __bf16 sm[2][8192];
  int tid = threadIdx.x, wave = tid >> 5, lane = tid & 31;
  int mtile0 = blockIdx.x * 8;
  int nt0    = blockIdx.y * 8;
  long abase = (long)mtile0 * 32 * 512;

  v8f acc[8];
  v8f zero = {0.f, 0.f, 0.f, 0.f, 0.f, 0.f, 0.f, 0.f};
#pragma unroll
  for (int t = 0; t < 8; ++t) acc[t] = zero;

  // A fragment base for a given global kt (kt<32 -> x, else msg).
  const __bf16* a0 = xp + abase;
  const __bf16* a1 = mp + abase;

  stage_tile(sm[0], a0, 32 * 512, wp + (long)nt0 * 512, tid);

  for (int kt = 0; kt < KT3; ++kt) {
    int cur = kt & 1;
    if (kt + 1 < KT3) {
      int kn = kt + 1;
      const __bf16* an = (kn < 32) ? (a0 + (long)kn * 512)
                                   : (a1 + (long)(kn - 32) * 512);
      stage_tile(sm[cur ^ 1], an, 32 * 512,
                 wp + ((long)kn * NT3 + nt0) * 512, tid);
    }
    async_wait4();
    __syncthreads();

    v16bf a = *(const v16bf*)(&sm[cur][wave * 512 + lane * 16]);
    v16bf bf[8];
#pragma unroll
    for (int t = 0; t < 8; ++t)
      bf[t] = *(const v16bf*)(&sm[cur][4096 + t * 512 + lane * 16]);
#pragma unroll
    for (int t = 0; t < 8; ++t)
      acc[t] = __builtin_amdgcn_wmma_f32_16x16x32_bf16(
          false, a, false, bf[t], (short)0, acc[t], false, false);

    __syncthreads();
  }

  int mb = (mtile0 + wave) * 16 + ((lane >> 4) << 3);
  int n = lane & 15;
#pragma unroll
  for (int t = 0; t < 8; ++t) {
    int col = (nt0 + t) * 16 + n;
    float bias = mbias[col];
#pragma unroll
    for (int r = 0; r < 8; ++r)
      out[(long)(mb + r) * Dn + col] = acc[t][r] + bias;
  }
}

// ----------------------------- LayerNorm -----------------------------------
__global__ __launch_bounds__(256) void k_ln(float* __restrict__ y,
                                            const float* __restrict__ gamma,
                                            const float* __restrict__ beta) {
  __shared__ float r1[8], r2[8], mv[2];
  long base = (long)blockIdx.x * Dn;
  int tid = threadIdx.x, lane = tid & 31, wave = tid >> 5;
  float v[4], s = 0.f, q = 0.f;
#pragma unroll
  for (int i = 0; i < 4; ++i) {
    v[i] = y[base + tid + i * 256];
    s += v[i];
    q += v[i] * v[i];
  }
#pragma unroll
  for (int off = 16; off > 0; off >>= 1) {
    s += __shfl_down(s, off, 32);
    q += __shfl_down(q, off, 32);
  }
  if (lane == 0) { r1[wave] = s; r2[wave] = q; }
  __syncthreads();
  if (tid == 0) {
    float ss = 0.f, qq = 0.f;
    for (int wv = 0; wv < 8; ++wv) { ss += r1[wv]; qq += r2[wv]; }
    float mu = ss * (1.f / Dn);
    float var = qq * (1.f / Dn) - mu * mu;
    mv[0] = mu;
    mv[1] = rsqrtf(var + 1e-5f);
  }
  __syncthreads();
  float mu = mv[0], rs = mv[1];
#pragma unroll
  for (int i = 0; i < 4; ++i) {
    int c = tid + i * 256;
    y[base + c] = (v[i] - mu) * rs * gamma[c] + beta[c];
  }
}

// ------------------------------ launcher -----------------------------------
extern "C" void kernel_launch(void* const* d_in, const int* in_sizes, int n_in,
                              void* d_out, int out_size, void* d_ws, size_t ws_size,
                              hipStream_t stream) {
  const float* x     = (const float*)d_in[0];
  const float* w1    = (const float*)d_in[1];
  const float* b1    = (const float*)d_in[2];
  const float* w2    = (const float*)d_in[3];
  const float* b2    = (const float*)d_in[4];
  const float* wv_w  = (const float*)d_in[5];
  const float* wv_b  = (const float*)d_in[6];
  const float* mw    = (const float*)d_in[7];
  const float* mb    = (const float*)d_in[8];
  const float* gamma = (const float*)d_in[9];
  const float* beta  = (const float*)d_in[10];
  float* out = (float*)d_out;

  char* ws = (char*)d_ws;
  __bf16* xp   = (__bf16*)(ws);                      // 16 MB
  __bf16* msgp = (__bf16*)(ws + (size_t)16777216);   // 16 MB
  __bf16* wcat = (__bf16*)(ws + (size_t)33554432);   //  4 MB
  __bf16* mwp  = (__bf16*)(ws + (size_t)37748736);   //  4 MB
  __bf16* c1   = (__bf16*)(ws + (size_t)41943040);   // 32 MB

  k_pack_x <<<32768, 256, 0, stream>>>(x, xp);
  k_pack_w <<<8192,  256, 0, stream>>>(w1, wv_w, wcat);
  k_pack_mw<<<8192,  256, 0, stream>>>(mw, mwp);
  k_gemm1  <<<dim3(64, 16), 256, 0, stream>>>(xp, wcat, c1);
  k_pass2  <<<8192,  256, 0, stream>>>(c1, b1, w2, b2, wv_b, msgp);
  k_gemm3  <<<dim3(64, 8), 256, 0, stream>>>(xp, msgp, mwp, mb, out);
  k_ln     <<<8192,  256, 0, stream>>>(out, gamma, beta);
}
